// FigLinear_29094108463113
// MI455X (gfx1250) — matrix operand order repaired and checked by reading, not compiled, not run on hardware
//
#include <hip/hip_runtime.h>
#include <hip/hip_bf16.h>

typedef __attribute__((ext_vector_type(16))) _Float16 v16h;
typedef __attribute__((ext_vector_type(8)))  _Float16 v8h;
typedef __attribute__((ext_vector_type(4)))  _Float16 v4h;
typedef __attribute__((ext_vector_type(8)))  float    v8f;
typedef __attribute__((ext_vector_type(4)))  float    v4f;
typedef __attribute__((ext_vector_type(4)))  int      v4i;

#define OUT_N 5632
#define IN_K  2048
#define M_TOK 8192
#define BM 128
#define BN 128
#define BK 32
#define LDT 40   // padded LDS row stride in halves (80B -> conflict-free b128 loads)

// A fragment 16x32 f16 (ISA 7.12.2): lane L -> row M = L&15;
// lanes 0-15 hold K 0..7 (v0-3) and 16..23 (v4-7); lanes 16-31 hold K 8..15, 24..31.
__device__ __forceinline__ v16h frag_a(const _Float16* t, int row0, int lane) {
  const _Float16* p = t + (row0 + (lane & 15)) * LDT + ((lane >> 4) << 3);
  v8h lo = *(const v8h*)p;
  v8h hi = *(const v8h*)(p + 16);
  v16h r;
#pragma unroll
  for (int i = 0; i < 8; ++i) { r[i] = lo[i]; r[i + 8] = hi[i]; }
  return r;
}

// B fragment 32x16 f16: lane L -> col N = L&15; lanes 0-15 hold K 0..7, 16..23; lanes 16-31 K 8..15, 24..31.
// LDS tile stored as [n][k] (K contiguous per output column == per W row).
__device__ __forceinline__ v16h frag_b(const _Float16* t, int col0, int lane) {
  const _Float16* p = t + (col0 + (lane & 15)) * LDT + ((lane >> 4) << 3);
  v8h lo = *(const v8h*)p;
  v8h hi = *(const v8h*)(p + 16);
  v16h r;
#pragma unroll
  for (int i = 0; i < 8; ++i) { r[i] = lo[i]; r[i + 8] = hi[i]; }
  return r;
}

__global__ __launch_bounds__(256)
void fused_q4_lora_gemm(const float* __restrict__ x,
                        const int*   __restrict__ packed,
                        const float* __restrict__ scales,
                        const float* __restrict__ zeros,
                        const float* __restrict__ lora_A,
                        const float* __restrict__ lora_B,
                        float* __restrict__ out) {
  __shared__ __align__(16) _Float16 As[BM * LDT];   // x tile (f32->f16), later LoRA partials
  __shared__ __align__(16) _Float16 Bs[BN * LDT];   // dequantized W tile, later lora_B tile
  __shared__ __align__(16) _Float16 ALs[16 * LDT];  // lora_A chunk transposed to [n=16][k=32]

  const int t    = threadIdx.x;
  const int lane = t & 31;
  const int wid  = t >> 5;
  const int wr   = wid >> 2;   // 0..1 : 64-row group
  const int wc   = wid & 3;    // 0..3 : 32-col group
  const int n0   = blockIdx.x * BN;
  const int m0   = blockIdx.y * BM;

  v8f acc[4][2];
  v8f lacc;
#pragma unroll
  for (int i = 0; i < 4; ++i)
#pragma unroll
    for (int j = 0; j < 2; ++j)
#pragma unroll
      for (int e = 0; e < 8; ++e) acc[i][j][e] = 0.0f;
#pragma unroll
  for (int e = 0; e < 8; ++e) lacc[e] = 0.0f;

  for (int k0 = 0; k0 < IN_K; k0 += BK) {
    // ---- stage x tile: 128x32 f32 -> f16 ----
#pragma unroll
    for (int p = 0; p < 4; ++p) {
      int row = p * 32 + (t >> 3);
      int kk  = (t & 7) << 2;
      v4f v = *(const v4f*)(x + (size_t)(m0 + row) * IN_K + k0 + kk);
      v4h h;
      h[0] = (_Float16)v[0]; h[1] = (_Float16)v[1];
      h[2] = (_Float16)v[2]; h[3] = (_Float16)v[3];
      *(v4h*)&As[row * LDT + kk] = h;
    }
    // ---- stage W tile: dequant int4 (one int32 per byte, low nibble first) ----
    {
      int n  = t >> 1;
      int kk = (t & 1) << 4;  // 0 or 16 (chunk of 16 weights = 8 packed ints)
      size_t widx = ((size_t)(n0 + n) * IN_K + (size_t)(k0 + kk)) >> 1;
      v4i p0 = *(const v4i*)(packed + widx);
      v4i p1 = *(const v4i*)(packed + widx + 4);
      int g = (n0 + n) * (IN_K / 128) + (k0 >> 7);
      float s = scales[g];
      float z = zeros[g];
      v8h w0, w1;
#pragma unroll
      for (int j = 0; j < 4; ++j) {
        int b0 = p0[j], b1 = p1[j];
        w0[2 * j]     = (_Float16)(((float)(b0 & 15)        - z) * s);
        w0[2 * j + 1] = (_Float16)(((float)((b0 >> 4) & 15) - z) * s);
        w1[2 * j]     = (_Float16)(((float)(b1 & 15)        - z) * s);
        w1[2 * j + 1] = (_Float16)(((float)((b1 >> 4) & 15) - z) * s);
      }
      *(v8h*)&Bs[n * LDT + kk]     = w0;
      *(v8h*)&Bs[n * LDT + kk + 8] = w1;
    }
    // ---- stage lora_A chunk [32,16] f32 -> ALs[n][k] f16 ----
#pragma unroll
    for (int i = 0; i < 2; ++i) {
      int e  = i * 256 + t;
      int kk = e >> 4;
      int n  = e & 15;
      ALs[n * LDT + kk] = (_Float16)lora_A[(size_t)(k0 + kk) * 16 + n];
    }
    __syncthreads();

    // ---- rank-16 path: each wave accumulates its 16-row stripe of (x @ A) ----
    {
      v16h la = frag_a(As, wid * 16, lane);
      v16h lb = frag_b(ALs, 0, lane);
      lacc = __builtin_amdgcn_wmma_f32_16x16x32_f16(false, la, false, lb,
                                                    (short)0, lacc, false, false);
    }
    // ---- main path: 4x2 WMMAs per wave ----
    v16h af[4];
#pragma unroll
    for (int i = 0; i < 4; ++i) af[i] = frag_a(As, wr * 64 + i * 16, lane);
    v16h bf[2];
#pragma unroll
    for (int j = 0; j < 2; ++j) bf[j] = frag_b(Bs, wc * 32 + j * 16, lane);
#pragma unroll
    for (int i = 0; i < 4; ++i)
#pragma unroll
      for (int j = 0; j < 2; ++j)
        acc[i][j] = __builtin_amdgcn_wmma_f32_16x16x32_f16(false, af[i], false, bf[j],
                                                           (short)0, acc[i][j], false, false);
    __syncthreads();
  }

  // ================= epilogue: fold LoRA term into accumulators =================
  // Write LA[128,16] partials (f32 C-layout -> f16) into As, zero-padded to K=32.
  {
    int rhalf = lane >> 4;
    int colc  = lane & 15;
#pragma unroll
    for (int r = 0; r < 8; ++r) {
      int row = wid * 16 + r + rhalf * 8;
      As[row * LDT + colc]      = (_Float16)lacc[r];
      As[row * LDT + 16 + colc] = (_Float16)0.0f;
    }
  }
  // Stage (2 * lora_B)[16,128] into Bs as [n][k], zero-padded to K=32.
#pragma unroll
  for (int i = 0; i < 8; ++i) {
    int e = i * 256 + t;
    int r = e >> 7;    // 0..15 rank index
    int n = e & 127;
    float v = 2.0f * lora_B[(size_t)r * OUT_N + n0 + n];
    Bs[n * LDT + r]      = (_Float16)v;
    Bs[n * LDT + 16 + r] = (_Float16)0.0f;
  }
  __syncthreads();

  // acc += LA x LB (K=16 zero-padded to 32)
#pragma unroll
  for (int i = 0; i < 4; ++i) {
    v16h af = frag_a(As, wr * 64 + i * 16, lane);
#pragma unroll
    for (int j = 0; j < 2; ++j) {
      v16h bf = frag_b(Bs, wc * 32 + j * 16, lane);
      acc[i][j] = __builtin_amdgcn_wmma_f32_16x16x32_f16(false, af, false, bf,
                                                         (short)0, acc[i][j], false, false);
    }
  }

  // ---- store C tile (f32) ----
  {
    int rhalf = lane >> 4;
    int colc  = lane & 15;
#pragma unroll
    for (int i = 0; i < 4; ++i)
#pragma unroll
      for (int j = 0; j < 2; ++j) {
        int col = n0 + wc * 32 + j * 16 + colc;
#pragma unroll
        for (int r = 0; r < 8; ++r) {
          int row = m0 + wr * 64 + i * 16 + r + rhalf * 8;
          out[(size_t)row * OUT_N + col] = acc[i][j][r];
        }
      }
  }
}

extern "C" void kernel_launch(void* const* d_in, const int* in_sizes, int n_in,
                              void* d_out, int out_size, void* d_ws, size_t ws_size,
                              hipStream_t stream) {
  (void)in_sizes; (void)n_in; (void)d_ws; (void)ws_size; (void)out_size;
  const float* x      = (const float*)d_in[0];
  const int*   packed = (const int*)  d_in[1];
  const float* scales = (const float*)d_in[2];
  const float* zeros  = (const float*)d_in[3];
  const float* lora_A = (const float*)d_in[4];
  const float* lora_B = (const float*)d_in[5];
  float* out = (float*)d_out;

  dim3 grid(OUT_N / BN, M_TOK / BM);   // 44 x 64
  fused_q4_lora_gemm<<<grid, 256, 0, stream>>>(x, packed, scales, zeros,
                                               lora_A, lora_B, out);
}
